// MSTDirectPredictor_3556232921531
// MI455X (gfx1250) — compile-verified, matrix operand order checked by reading, not run on hardware
//
#include <hip/hip_runtime.h>
#include <hip/hip_bf16.h>

#define EPSV 1e-5f

typedef __bf16 v16bf __attribute__((ext_vector_type(16)));
typedef float  v8f   __attribute__((ext_vector_type(8)));

union AF { v16bf v; unsigned short u[16]; unsigned w[8]; uint4 q[2]; };

// ---------- helpers ----------
static __device__ inline unsigned short f2bfu(float f) {
  union { float f; unsigned u; } x; x.f = f;
  unsigned r = x.u + 0x7FFFu + ((x.u >> 16) & 1u);   // RNE truncate to bf16
  return (unsigned short)(r >> 16);
}

// pack two f32 -> packed bf16x2 dword (hardware v_cvt_pk_bf16_f32 when available)
#if __has_builtin(__builtin_amdgcn_cvt_pk_bf16_f32)
static __device__ inline unsigned pk2(float a, float b) {
  auto t = __builtin_amdgcn_cvt_pk_bf16_f32(a, b);
  unsigned r; __builtin_memcpy(&r, &t, 4); return r;
}
#else
static __device__ inline unsigned pk2(float a, float b) {
  return (unsigned)f2bfu(a) | ((unsigned)f2bfu(b) << 16);
}
#endif

// order-preserving float<->uint encoding (for hardware atomic max), branchless
static __device__ inline unsigned encF(float f) {
  unsigned u = __float_as_uint(f);
  return u ^ ((unsigned)((int)u >> 31) | 0x80000000u);
}
static __device__ inline float decF(unsigned u) {
  unsigned m = (~(unsigned)((int)u >> 31)) | 0x80000000u;
  return __uint_as_float(u ^ m);
}

static __device__ inline v8f wmma_bf16(const AF& a, const AF& b, v8f c) {
  // D = A(16x32 bf16) * B(32x16 bf16) + C(16x16 f32)
  return __builtin_amdgcn_wmma_f32_16x16x32_bf16(false, a.v, false, b.v,
                                                 (short)0, c, false, false);
}

// pack 8 contiguous f32 -> 4 packed bf16 dwords
static __device__ inline void pack8w(unsigned* w, const float* p) {
  float4 x0 = *(const float4*)p, x1 = *(const float4*)(p + 4);
  w[0] = pk2(x0.x, x0.y); w[1] = pk2(x0.z, x0.w);
  w[2] = pk2(x1.x, x1.y); w[3] = pk2(x1.z, x1.w);
}

// pack 8 contiguous relu(a+b) -> 4 packed bf16 dwords
static __device__ inline void pack8w_relu_sum(unsigned* w, const float* a, const float* b) {
  float4 x0 = *(const float4*)a, x1 = *(const float4*)(a + 4);
  float4 y0 = *(const float4*)b, y1 = *(const float4*)(b + 4);
  w[0] = pk2(fmaxf(x0.x + y0.x, 0.f), fmaxf(x0.y + y0.y, 0.f));
  w[1] = pk2(fmaxf(x0.z + y0.z, 0.f), fmaxf(x0.w + y0.w, 0.f));
  w[2] = pk2(fmaxf(x1.x + y1.x, 0.f), fmaxf(x1.y + y1.y, 0.f));
  w[3] = pk2(fmaxf(x1.z + y1.z, 0.f), fmaxf(x1.w + y1.w, 0.f));
}

// A fragment (16x32) from row-major f32, vectorized, zero-pad cols >= Ka (Ka multiple of 8)
static __device__ inline void load_a_f32v(AF& f, const float* __restrict__ A,
                                          int lda, int Ka, int row0, int k0, int lane) {
  int r  = row0 + (lane & 15);
  int kb = k0 + ((lane >> 4) << 3);
  const float* p = A + (size_t)r * lda + kb;
  if (kb + 8 <= Ka) pack8w(&f.w[0], p);
  else { f.q[0].x = 0; f.q[0].y = 0; f.q[0].z = 0; f.q[0].w = 0; }
  if (kb + 24 <= Ka) pack8w(&f.w[4], p + 16);
  else { f.q[1].x = 0; f.q[1].y = 0; f.q[1].z = 0; f.q[1].w = 0; }
}

// B fragment from pre-shuffled fragment-ready weight buffer: 2 x 128-bit loads
static __device__ inline void load_b_shuf(AF& f, const unsigned short* __restrict__ W,
                                          int numKc, int kc, int nc, int lane) {
  const uint4* p = (const uint4*)(W + (((size_t)nc * numKc + kc) * 32 + lane) * 16);
  f.q[0] = p[0];
  f.q[1] = p[1];
}

// A fragment from fragment-ready LDS staging: 2 x ds_load_b128
static __device__ inline void load_a_frag(AF& f, const unsigned short* S, int kc, int lane) {
  const uint4* p = (const uint4*)(S + kc * 512 + lane * 16);
  f.q[0] = p[0];
  f.q[1] = p[1];
}

// position of element (row r, col k) inside fragment-ready layout (512 ushorts per 32-k chunk)
static __device__ inline int frag_pos(int r, int k) {
  int kc = k >> 5, kk = k & 31;
  int q = kk >> 3;                  // 0..3
  int h = q & 1;                    // half-wave
  int j = (kk & 7) + ((q >> 1) << 3);
  return kc * 512 + (h * 16 + r) * 16 + j;
}

static __device__ inline float lrelu(float x) { return x > 0.f ? x : 0.2f * x; }

// ---------- elementwise / setup kernels ----------
__global__ void k_fill(float* p, long n, float v) {
  long i = (long)blockIdx.x * blockDim.x + threadIdx.x;
  if (i < n) p[i] = v;
}

__global__ void k_fill_enc(unsigned* p, long n, float v) {
  long i = (long)blockIdx.x * blockDim.x + threadIdx.x;
  if (i < n) p[i] = encF(v);
}

// shuffle a Kp x N f32 weight into fragment-ready bf16 layout (zero rows >= Ka)
__global__ void k_shufw(const float* __restrict__ W, unsigned short* __restrict__ D,
                        int N, int numKc, int Ka, long total) {
  long i = (long)blockIdx.x * blockDim.x + threadIdx.x;
  if (i >= total) return;
  int j = (int)(i & 15);
  int lane = (int)((i >> 4) & 31);
  int frag = (int)(i >> 9);
  int kc = frag % numKc, nc = frag / numKc;
  int c = nc * 16 + (lane & 15);
  int kb = kc * 32 + ((lane >> 4) << 3);
  int k = kb + ((j < 8) ? j : (8 + j));
  float v = (k < Ka) ? W[(size_t)k * N + c] : 0.f;
  D[i] = f2bfu(v);
}

// EdgeConv weight prep, fragment-ready: Wd = W1[0:128]-W1[128:256], Ws = W1[128:256]
__global__ void k_shufw_ec(const float* __restrict__ W1, unsigned short* __restrict__ Dd,
                           unsigned short* __restrict__ Ds) {
  long i = (long)blockIdx.x * blockDim.x + threadIdx.x;
  if (i >= 128 * 128) return;
  int j = (int)(i & 15);
  int lane = (int)((i >> 4) & 31);
  int frag = (int)(i >> 9);
  int kc = frag & 3, nc = frag >> 2;           // numKc = 4
  int c = nc * 16 + (lane & 15);
  int kb = kc * 32 + ((lane >> 4) << 3);
  int k = kb + ((j < 8) ? j : (8 + j));
  float a = W1[(size_t)k * 128 + c];
  float b = W1[(size_t)(128 + k) * 128 + c];
  Dd[i] = f2bfu(a - b);
  Ds[i] = f2bfu(b);
}

__global__ void k_deg(const int* __restrict__ dst, float* __restrict__ deg, int E) {
  int i = blockIdx.x * blockDim.x + threadIdx.x;
  if (i < E) atomicAdd(&deg[dst[i]], 1.0f);
}

__global__ void k_dfin(const float* __restrict__ deg, float* __restrict__ dinv,
                       float* __restrict__ invdeg, int N) {
  int i = blockIdx.x * blockDim.x + threadIdx.x;
  if (i < N) { float d = deg[i] + 1.0f; dinv[i] = rsqrtf(d); invdeg[i] = 1.0f / d; }
}

// ---------- generic WMMA GEMM: C[M x Nout] = act(A[M x Ka] @ Wshuf[Kp x Nout] + bias)
template <int NK>
__global__ __launch_bounds__(32) void k_gemm(const float* __restrict__ A, int lda, int Ka,
                                             const unsigned short* __restrict__ W, int Nout,
                                             const float* __restrict__ bias,
                                             float* __restrict__ C, int act) {
  int lane = threadIdx.x;
  int row0 = blockIdx.x * 16;
  __builtin_prefetch(A + (size_t)row0 * lda, 0, 1);
  AF afr[NK];
#pragma unroll
  for (int kc = 0; kc < NK; ++kc) load_a_f32v(afr[kc], A, lda, Ka, row0, kc * 32, lane);
  int c15 = lane & 15, hi = (lane >> 4) << 3;
  for (int nc = 0; nc < Nout; nc += 16) {
    v8f acc = {};
#pragma unroll
    for (int kc = 0; kc < NK; ++kc) {
      AF bfr; load_b_shuf(bfr, W, NK, kc, nc >> 4, lane);
      acc = wmma_bf16(afr[kc], bfr, acc);
    }
    int col = nc + c15;
    float bb = bias ? bias[col] : 0.0f;
#pragma unroll
    for (int i = 0; i < 8; ++i) {
      float v = acc[i] + bb;
      if (act == 1) v = fmaxf(v, 0.0f);
      C[(size_t)(row0 + i + hi) * Nout + col] = v;
    }
  }
}

// ---------- BatchNorm ----------
__global__ void k_bn_stats(const float* __restrict__ A, int rows, float* __restrict__ stats) {
  int c = threadIdx.x;                 // 128 threads = one per channel
  int r0 = blockIdx.x * 256;
  float s = 0.f, ss = 0.f;
  for (int i = 0; i < 256; ++i) {
    int r = r0 + i;
    if (r < rows) { float v = A[(size_t)r * 128 + c]; s += v; ss += v * v; }
  }
  atomicAdd(&stats[c], s);
  atomicAdd(&stats[128 + c], ss);
}

__global__ void k_bn_apply(const float* __restrict__ hn, const float* __restrict__ stats,
                           const float* __restrict__ g, const float* __restrict__ be,
                           float* __restrict__ h, float invN, long n) {
  long i = (long)blockIdx.x * blockDim.x + threadIdx.x;
  if (i < n) {
    int c = (int)(i & 127);
    float mu = stats[c] * invN;
    float var = stats[128 + c] * invN - mu * mu;
    float v = (hn[i] - mu) * rsqrtf(var + EPSV) * g[c] + be[c];
    h[i] = fmaxf(v, 0.0f);
  }
}

// h = relu(hn) (+ hn if res)
__global__ void k_act(const float* __restrict__ hn, float* __restrict__ h, long n, int res) {
  long i = (long)blockIdx.x * blockDim.x + threadIdx.x;
  if (i < n) { float v = hn[i]; h[i] = fmaxf(v, 0.0f) + (res ? v : 0.0f); }
}

// EdgeConv finalize: decode atomic-max encoding, clean empty segments, act/res
__global__ void k_act_ec(const unsigned* __restrict__ agg, float* __restrict__ h, long n, int res) {
  long i = (long)blockIdx.x * blockDim.x + threadIdx.x;
  if (i < n) {
    float v = decF(agg[i]);
    v = (v < -1.0e37f) ? 0.0f : v;
    h[i] = fmaxf(v, 0.0f) + (res ? v : 0.0f);
  }
}

// ---------- GCN ----------
__global__ void k_gcn_scatter(const float* __restrict__ xw, const int* __restrict__ src,
                              const int* __restrict__ dst, const float* __restrict__ dinv,
                              float* __restrict__ agg, int E) {
  long i = (long)blockIdx.x * blockDim.x + threadIdx.x;
  if (i >= (long)E * 32) return;
  int e = (int)(i >> 5), t = (int)(i & 31);
  int s = src[e], d = dst[e];
  float nrm = dinv[s] * dinv[d];
  const float4 v = *(const float4*)&xw[(size_t)s * 128 + t * 4];
  float* o = &agg[(size_t)d * 128 + t * 4];
  atomicAdd(o + 0, v.x * nrm);
  atomicAdd(o + 1, v.y * nrm);
  atomicAdd(o + 2, v.z * nrm);
  atomicAdd(o + 3, v.w * nrm);
}

__global__ void k_gcn_self(const float* __restrict__ agg, const float* __restrict__ xw,
                           const float* __restrict__ invdeg, const float* __restrict__ b,
                           float* __restrict__ hn, long n) {
  long i = (long)blockIdx.x * blockDim.x + threadIdx.x;
  if (i < n) {
    int nd = (int)(i >> 7), c = (int)(i & 127);
    hn[i] = agg[i] + xw[i] * invdeg[nd] + b[c];
  }
}

// ---------- GAT ----------
__global__ void k_gat_al(const float* __restrict__ xw, const float* __restrict__ as,
                         const float* __restrict__ ad, float* __restrict__ als,
                         float* __restrict__ ald, int N) {
  int i = blockIdx.x * blockDim.x + threadIdx.x;
  if (i >= N * 8) return;
  int n = i >> 3, hd = i & 7;
  float s = 0.f, d = 0.f;
  const float* p = &xw[(size_t)n * 128 + hd * 16];
#pragma unroll
  for (int c = 0; c < 16; ++c) { float v = p[c]; s += v * as[hd * 16 + c]; d += v * ad[hd * 16 + c]; }
  als[i] = s; ald[i] = d;
}

__global__ void k_gat_selfinit(const float* __restrict__ als, const float* __restrict__ ald,
                               unsigned* __restrict__ emax, int N) {
  int i = blockIdx.x * blockDim.x + threadIdx.x;
  if (i < N * 8) emax[i] = encF(lrelu(als[i] + ald[i]));
}

__global__ void k_gat_emax(const int* __restrict__ src, const int* __restrict__ dst,
                           const float* __restrict__ als, const float* __restrict__ ald,
                           unsigned* __restrict__ emax, int E) {
  long i = (long)blockIdx.x * blockDim.x + threadIdx.x;
  if (i >= (long)E * 8) return;
  int e = (int)(i >> 3), hd = (int)(i & 7);
  float v = lrelu(als[src[e] * 8 + hd] + ald[dst[e] * 8 + hd]);
  atomicMax(&emax[dst[e] * 8 + hd], encF(v));
}

__global__ void k_gat_denominit(const float* __restrict__ als, const float* __restrict__ ald,
                                const unsigned* __restrict__ emax, float* __restrict__ denom, int N) {
  int i = blockIdx.x * blockDim.x + threadIdx.x;
  if (i < N * 8) denom[i] = expf(lrelu(als[i] + ald[i]) - decF(emax[i]));
}

__global__ void k_gat_denom(const int* __restrict__ src, const int* __restrict__ dst,
                            const float* __restrict__ als, const float* __restrict__ ald,
                            const unsigned* __restrict__ emax, float* __restrict__ denom, int E) {
  long i = (long)blockIdx.x * blockDim.x + threadIdx.x;
  if (i >= (long)E * 8) return;
  int e = (int)(i >> 3), hd = (int)(i & 7);
  int d8 = dst[e] * 8 + hd;
  float v = lrelu(als[src[e] * 8 + hd] + ald[d8]);
  atomicAdd(&denom[d8], expf(v - decF(emax[d8])));
}

__global__ void k_gat_outinit(const float* __restrict__ xw, const float* __restrict__ als,
                              const float* __restrict__ ald, const unsigned* __restrict__ emax,
                              const float* __restrict__ denom, const float* __restrict__ b,
                              float* __restrict__ hn, long n) {
  long i = (long)blockIdx.x * blockDim.x + threadIdx.x;
  if (i < n) {
    int nd = (int)(i >> 7), c = (int)(i & 127), hd = c >> 4;
    int n8 = nd * 8 + hd;
    float alpha = expf(lrelu(als[n8] + ald[n8]) - decF(emax[n8])) / denom[n8];
    hn[i] = xw[i] * alpha + b[c];
  }
}

__global__ void k_gat_scatter(const float* __restrict__ xw, const int* __restrict__ src,
                              const int* __restrict__ dst, const float* __restrict__ als,
                              const float* __restrict__ ald, const unsigned* __restrict__ emax,
                              const float* __restrict__ denom, float* __restrict__ hn, int E) {
  long i = (long)blockIdx.x * blockDim.x + threadIdx.x;
  if (i >= (long)E * 32) return;
  int e = (int)(i >> 5), t = (int)(i & 31);
  int s = src[e], d = dst[e];
  int hd = t >> 2;
  int d8 = d * 8 + hd;
  float w = expf(lrelu(als[s * 8 + hd] + ald[d8]) - decF(emax[d8])) / denom[d8];
  const float4 v = *(const float4*)&xw[(size_t)s * 128 + t * 4];
  float* o = &hn[(size_t)d * 128 + t * 4];
  atomicAdd(o + 0, v.x * w);
  atomicAdd(o + 1, v.y * w);
  atomicAdd(o + 2, v.z * w);
  atomicAdd(o + 3, v.w * w);
}

// ---------- EdgeConv fused edge kernel: m = relu(hA[dst]+hB[src]); agg[dst] = max(agg, m@W2 + b2)
__global__ __launch_bounds__(32) void k_ec_edge(const float* __restrict__ hA,
                                                const float* __restrict__ hB,
                                                const int* __restrict__ src,
                                                const int* __restrict__ dst,
                                                const unsigned short* __restrict__ W2,
                                                const float* __restrict__ b2,
                                                unsigned* __restrict__ agg) {
  int lane = threadIdx.x;
  int e0 = blockIdx.x * 16;
  int r = lane & 15, kb = (lane >> 4) << 3, hi = (lane >> 4) << 3, c15 = lane & 15;
  int e = e0 + r;
  int dval = dst[e];                     // lane r holds dst of tile-row r
  const float* pa = hA + (size_t)dval * 128;
  const float* pb = hB + (size_t)src[e] * 128;
  AF m[4];
#pragma unroll
  for (int kc = 0; kc < 4; ++kc) {
    const float* qa = pa + kc * 32 + kb;
    const float* qb = pb + kc * 32 + kb;
    pack8w_relu_sum(&m[kc].w[0], qa, qb);
    pack8w_relu_sum(&m[kc].w[4], qa + 16, qb + 16);
  }
  for (int nc = 0; nc < 128; nc += 16) {
    v8f acc = {};
#pragma unroll
    for (int kc = 0; kc < 4; ++kc) {
      AF bfr; load_b_shuf(bfr, W2, 4, kc, nc >> 4, lane);
      acc = wmma_bf16(m[kc], bfr, acc);
    }
    int col = nc + c15;
    float bb = b2[col];
#pragma unroll
    for (int i = 0; i < 8; ++i) {
      int drow = __shfl(dval, i + hi, 32);   // dst of output row i+hi
      atomicMax(&agg[(size_t)drow * 128 + col], encF(acc[i] + bb));
    }
  }
}

// ---------- graph context: g = mean(h); g2 = relu(g@W1+b1)@W2+b2; gvec = g2@epW1[384:512] + ep_b1
__global__ void k_gc(const float* __restrict__ gsum, float invN,
                     const float* __restrict__ W1, const float* __restrict__ b1,
                     const float* __restrict__ W2, const float* __restrict__ b2,
                     const float* __restrict__ epW1, const float* __restrict__ epb1,
                     float* __restrict__ gvec) {
  __shared__ float g[128], t[128], h2[128];
  int tid = threadIdx.x;  // 256
  if (tid < 128) g[tid] = gsum[tid] * invN;
  __syncthreads();
  if (tid < 128) {
    float s = b1[tid];
    for (int k = 0; k < 128; ++k) s += g[k] * W1[k * 128 + tid];
    t[tid] = fmaxf(s, 0.0f);
  }
  __syncthreads();
  if (tid < 128) {
    float s = b2[tid];
    for (int k = 0; k < 128; ++k) s += t[k] * W2[k * 128 + tid];
    h2[tid] = s;
  }
  __syncthreads();
  {
    float s = epb1[tid];
    for (int k = 0; k < 128; ++k) s += h2[k] * epW1[(384 + k) * 256 + tid];
    gvec[tid] = s;
  }
}

// ---------- fused edge predictor:
// eh = LN(relu(ea@eeW+eeb));  z = LN(relu(eh@W1c + u1[src] + u2[dst] + gvec));
// y  = LN(relu(z@W2+b2));     out = y@w3 + b3
__global__ __launch_bounds__(32) void k_ep(const float* __restrict__ ea,
    const int* __restrict__ src, const int* __restrict__ dst,
    const unsigned short* __restrict__ eeW, const float* __restrict__ eeB,
    const float* __restrict__ eeG, const float* __restrict__ eeBe,
    const unsigned short* __restrict__ W1c,
    const float* __restrict__ u1, const float* __restrict__ u2,
    const float* __restrict__ gvec,
    const float* __restrict__ g1, const float* __restrict__ be1,
    const unsigned short* __restrict__ W2, const float* __restrict__ b2,
    const float* __restrict__ g2, const float* __restrict__ be2,
    const float* __restrict__ w3, const float* __restrict__ b3,
    float* __restrict__ out) {
  __shared__ alignas(16) float sf[16 * 256];           // staging f32 (eh/z/y tiles)
  __shared__ alignas(16) unsigned short sz[8 * 512];   // z bf16, fragment-ready (K=256)
  __shared__ alignas(16) unsigned short se[4 * 512];   // eh bf16, fragment-ready (K=128)
  __shared__ int s_src[16], s_dst[16];
  int lane = threadIdx.x;
  int e0 = blockIdx.x * 16;
  int r = lane & 15, kb = (lane >> 4) << 3, hi = (lane >> 4) << 3, c15 = lane & 15;
  int half = lane >> 4;

  if (lane < 16) { s_src[lane] = src[e0 + lane]; s_dst[lane] = dst[e0 + lane]; }

  // full-wave LayerNorm over sf rows (lane pairs l,l+16 split each row), store
  // normalized bf16 into fragment-ready LDS buffer as packed dwords
  auto ln_store = [&](int width, const float* gam, const float* bet, unsigned short* dstS) {
    int hw = width >> 1;
    const float* row = sf + r * width + half * hw;
    int cbase = half * hw;
    float s = 0.f, ss = 0.f;
    for (int c = 0; c < hw; ++c) { float v = row[c]; s += v; ss += v * v; }
    s += __shfl_xor(s, 16, 32);
    ss += __shfl_xor(ss, 16, 32);
    float mu = s / (float)width, var = ss / (float)width - mu * mu;
    float rsv = rsqrtf(var + EPSV);
    for (int c = 0; c < hw; c += 2) {
      int cc = cbase + c;
      float v0 = (row[c]     - mu) * rsv * gam[cc]     + bet[cc];
      float v1 = (row[c + 1] - mu) * rsv * gam[cc + 1] + bet[cc + 1];
      *(unsigned*)&dstS[frag_pos(r, cc)] = pk2(v0, v1);
    }
  };

  // ---- stage 1: eh raw = relu(ea @ eeW + eeB), K = 16 padded to 32
  AF af;
  {
    const float* p = ea + (size_t)(e0 + r) * 16 + kb;
    if (kb + 8 <= 16) pack8w(&af.w[0], p);
    else { af.q[0].x = 0; af.q[0].y = 0; af.q[0].z = 0; af.q[0].w = 0; }
    af.q[1].x = 0; af.q[1].y = 0; af.q[1].z = 0; af.q[1].w = 0;   // k >= 16 is pad
  }
  for (int nc = 0; nc < 128; nc += 16) {
    AF bfr; load_b_shuf(bfr, eeW, 1, 0, nc >> 4, lane);
    v8f z0 = {};
    v8f acc = wmma_bf16(af, bfr, z0);
    int col = nc + c15;
#pragma unroll
    for (int i = 0; i < 8; ++i) sf[(i + hi) * 128 + col] = fmaxf(acc[i] + eeB[col], 0.0f);
  }
  __syncthreads();
  ln_store(128, eeG, eeBe, se);
  __syncthreads();

  // ---- stage u: sf[row, 0:256] = u1[src[row]] + u2[dst[row]] + gvec (coalesced float4)
  {
    float gv[8];
#pragma unroll
    for (int j = 0; j < 8; ++j) gv[j] = gvec[lane * 8 + j];
    for (int rr = 0; rr < 16; ++rr) {
      const float* p1 = u1 + (size_t)s_src[rr] * 256 + lane * 8;
      const float* p2 = u2 + (size_t)s_dst[rr] * 256 + lane * 8;
      float4 a0 = ((const float4*)p1)[0], a1 = ((const float4*)p1)[1];
      float4 b0 = ((const float4*)p2)[0], b1 = ((const float4*)p2)[1];
      float* o = sf + rr * 256 + lane * 8;
      o[0] = a0.x + b0.x + gv[0]; o[1] = a0.y + b0.y + gv[1];
      o[2] = a0.z + b0.z + gv[2]; o[3] = a0.w + b0.w + gv[3];
      o[4] = a1.x + b1.x + gv[4]; o[5] = a1.y + b1.y + gv[5];
      o[6] = a1.z + b1.z + gv[6]; o[7] = a1.w + b1.w + gv[7];
    }
  }
  __syncthreads();

  // ---- stage 2: z = relu(eh @ W1c + staged), N=256, K=128
  AF ef[4];
#pragma unroll
  for (int kc = 0; kc < 4; ++kc) load_a_frag(ef[kc], se, kc, lane);
  for (int nc = 0; nc < 256; nc += 16) {
    v8f acc = {};
#pragma unroll
    for (int kc = 0; kc < 4; ++kc) {
      AF bfr; load_b_shuf(bfr, W1c, 4, kc, nc >> 4, lane);
      acc = wmma_bf16(ef[kc], bfr, acc);
    }
    int col = nc + c15;
#pragma unroll
    for (int i = 0; i < 8; ++i) {
      int idx = (i + hi) * 256 + col;
      sf[idx] = fmaxf(acc[i] + sf[idx], 0.0f);   // same-lane read-then-write
    }
  }
  __syncthreads();
  ln_store(256, g1, be1, sz);
  __syncthreads();

  // ---- stage 3: y = relu(z @ W2 + b2), N=128, K=256
  AF zf[8];
#pragma unroll
  for (int kc = 0; kc < 8; ++kc) load_a_frag(zf[kc], sz, kc, lane);
  __syncthreads();
  for (int nc = 0; nc < 128; nc += 16) {
    v8f acc = {};
#pragma unroll
    for (int kc = 0; kc < 8; ++kc) {
      AF bfr; load_b_shuf(bfr, W2, 8, kc, nc >> 4, lane);
      acc = wmma_bf16(zf[kc], bfr, acc);
    }
    int col = nc + c15;
#pragma unroll
    for (int i = 0; i < 8; ++i) sf[(i + hi) * 128 + col] = fmaxf(acc[i] + b2[col], 0.0f);
  }
  __syncthreads();

  // ---- stage 4: full-wave LN(128) + dot with w3
  {
    const float* row = sf + r * 128 + half * 64;
    int cbase = half * 64;
    float s = 0.f, ss = 0.f;
    for (int c = 0; c < 64; ++c) { float v = row[c]; s += v; ss += v * v; }
    s += __shfl_xor(s, 16, 32);
    ss += __shfl_xor(ss, 16, 32);
    float mu = s * (1.0f / 128.0f), var = ss * (1.0f / 128.0f) - mu * mu;
    float rsv = rsqrtf(var + EPSV);
    float acc2 = 0.f;
    for (int c = 0; c < 64; ++c)
      acc2 += ((row[c] - mu) * rsv * g2[cbase + c] + be2[cbase + c]) * w3[cbase + c];
    acc2 += __shfl_xor(acc2, 16, 32);
    if (lane < 16) out[e0 + r] = acc2 + b3[0];
  }
}

// =====================================================================
extern "C" void kernel_launch(void* const* d_in, const int* in_sizes, int n_in,
                              void* d_out, int out_size, void* d_ws, size_t ws_size,
                              hipStream_t stream) {
  const int N = in_sizes[0] / 32;   // 50000
  const int E = in_sizes[2] / 16;   // 800000

  const float* x  = (const float*)d_in[0];
  const int*   ei = (const int*)d_in[1];
  const float* ea = (const float*)d_in[2];
  const int* srcI = ei;
  const int* dstI = ei + E;

  const float* neW  = (const float*)d_in[3];
  const float* neB  = (const float*)d_in[4];
  const float* neG  = (const float*)d_in[5];
  const float* neBe = (const float*)d_in[6];
  const float* eeW  = (const float*)d_in[7];
  const float* eeB  = (const float*)d_in[8];
  const float* eeG  = (const float*)d_in[9];
  const float* eeBe = (const float*)d_in[10];
  const float* gcn0W = (const float*)d_in[11];
  const float* gcn0B = (const float*)d_in[12];
  const float* gcn3W = (const float*)d_in[13];
  const float* gcn3B = (const float*)d_in[14];
  const float* gat1W = (const float*)d_in[15];
  const float* gat1As = (const float*)d_in[16];
  const float* gat1Ad = (const float*)d_in[17];
  const float* gat1B = (const float*)d_in[18];
  const float* gat4W = (const float*)d_in[19];
  const float* gat4As = (const float*)d_in[20];
  const float* gat4Ad = (const float*)d_in[21];
  const float* gat4B = (const float*)d_in[22];
  const float* ec2W1 = (const float*)d_in[23];
  const float* ec2B1 = (const float*)d_in[24];
  const float* ec2W2 = (const float*)d_in[25];
  const float* ec2B2 = (const float*)d_in[26];
  const float* ec5W1 = (const float*)d_in[27];
  const float* ec5B1 = (const float*)d_in[28];
  const float* ec5W2 = (const float*)d_in[29];
  const float* ec5B2 = (const float*)d_in[30];
  const float* gcW1 = (const float*)d_in[31];
  const float* gcB1 = (const float*)d_in[32];
  const float* gcW2 = (const float*)d_in[33];
  const float* gcB2 = (const float*)d_in[34];
  const float* epW1 = (const float*)d_in[35];
  const float* epB1 = (const float*)d_in[36];
  const float* epG1 = (const float*)d_in[37];
  const float* epBe1 = (const float*)d_in[38];
  const float* epW2 = (const float*)d_in[39];
  const float* epB2 = (const float*)d_in[40];
  const float* epG2 = (const float*)d_in[41];
  const float* epBe2 = (const float*)d_in[42];
  const float* epW3 = (const float*)d_in[43];
  const float* epB3 = (const float*)d_in[44];

  // ---- workspace bump allocator
  char* wsp = (char*)d_ws;
  auto alloc = [&](size_t bytes) -> void* {
    void* p = wsp;
    wsp += (bytes + 255) & ~(size_t)255;
    return p;
  };
  float* h    = (float*)alloc((size_t)N * 128 * 4);
  float* hn   = (float*)alloc((size_t)N * 128 * 4);
  float* xw   = (float*)alloc((size_t)N * 128 * 4);
  float* agg  = (float*)alloc((size_t)N * 128 * 4);   // f32 (GCN) or encoded u32 (EC)
  float* hA   = (float*)alloc((size_t)N * 128 * 4);
  float* hB   = (float*)alloc((size_t)N * 128 * 4);
  float* u1   = (float*)alloc((size_t)N * 256 * 4);
  float* u2   = (float*)alloc((size_t)N * 256 * 4);
  float* deg  = (float*)alloc((size_t)N * 4);
  float* dinv = (float*)alloc((size_t)N * 4);
  float* invd = (float*)alloc((size_t)N * 4);
  float* als  = (float*)alloc((size_t)N * 8 * 4);
  float* ald  = (float*)alloc((size_t)N * 8 * 4);
  unsigned* emax = (unsigned*)alloc((size_t)N * 8 * 4);
  float* denom= (float*)alloc((size_t)N * 8 * 4);
  float* stats= (float*)alloc(256 * 4);
  float* gvec = (float*)alloc(256 * 4);
  unsigned short* neWb   = (unsigned short*)alloc(32 * 128 * 2);
  unsigned short* eeWb   = (unsigned short*)alloc(32 * 128 * 2);
  unsigned short* gcn0Wb = (unsigned short*)alloc(128 * 128 * 2);
  unsigned short* gcn3Wb = (unsigned short*)alloc(128 * 128 * 2);
  unsigned short* gat1Wb = (unsigned short*)alloc(128 * 128 * 2);
  unsigned short* gat4Wb = (unsigned short*)alloc(128 * 128 * 2);
  unsigned short* ec2Wd  = (unsigned short*)alloc(128 * 128 * 2);
  unsigned short* ec2Ws  = (unsigned short*)alloc(128 * 128 * 2);
  unsigned short* ec2W2b = (unsigned short*)alloc(128 * 128 * 2);
  unsigned short* ec5Wd  = (unsigned short*)alloc(128 * 128 * 2);
  unsigned short* ec5Ws  = (unsigned short*)alloc(128 * 128 * 2);
  unsigned short* ec5W2b = (unsigned short*)alloc(128 * 128 * 2);
  unsigned short* epW1r0 = (unsigned short*)alloc(128 * 256 * 2);
  unsigned short* epW1r1 = (unsigned short*)alloc(128 * 256 * 2);
  unsigned short* epW1r2 = (unsigned short*)alloc(128 * 256 * 2);
  unsigned short* epW2b  = (unsigned short*)alloc(256 * 128 * 2);

  auto g1 = [](long n) { return dim3((unsigned)((n + 255) / 256)); };

  // ---- weight conversion to fragment-ready bf16 layouts
  k_shufw<<<g1(4096), 256, 0, stream>>>(neW, neWb, 128, 1, 32, 4096);
  k_shufw<<<g1(4096), 256, 0, stream>>>(eeW, eeWb, 128, 1, 16, 4096);
  k_shufw<<<g1(16384), 256, 0, stream>>>(gcn0W, gcn0Wb, 128, 4, 128, 16384);
  k_shufw<<<g1(16384), 256, 0, stream>>>(gcn3W, gcn3Wb, 128, 4, 128, 16384);
  k_shufw<<<g1(16384), 256, 0, stream>>>(gat1W, gat1Wb, 128, 4, 128, 16384);
  k_shufw<<<g1(16384), 256, 0, stream>>>(gat4W, gat4Wb, 128, 4, 128, 16384);
  k_shufw_ec<<<g1(16384), 256, 0, stream>>>(ec2W1, ec2Wd, ec2Ws);
  k_shufw<<<g1(16384), 256, 0, stream>>>(ec2W2, ec2W2b, 128, 4, 128, 16384);
  k_shufw_ec<<<g1(16384), 256, 0, stream>>>(ec5W1, ec5Wd, ec5Ws);
  k_shufw<<<g1(16384), 256, 0, stream>>>(ec5W2, ec5W2b, 128, 4, 128, 16384);
  k_shufw<<<g1(32768), 256, 0, stream>>>(epW1, epW1r0, 256, 4, 128, 32768);
  k_shufw<<<g1(32768), 256, 0, stream>>>(epW1 + 128 * 256, epW1r1, 256, 4, 128, 32768);
  k_shufw<<<g1(32768), 256, 0, stream>>>(epW1 + 256 * 256, epW1r2, 256, 4, 128, 32768);
  k_shufw<<<g1(32768), 256, 0, stream>>>(epW2, epW2b, 128, 8, 256, 32768);

  // ---- degrees
  k_fill<<<g1(N), 256, 0, stream>>>(deg, N, 0.f);
  k_deg<<<g1(E), 256, 0, stream>>>(dstI, deg, E);
  k_dfin<<<g1(N), 256, 0, stream>>>(deg, dinv, invd, N);

  const long NH = (long)N * 128;

  // ---- node encoder: Linear -> BN(batch stats) -> ReLU
  k_gemm<1><<<N / 16, 32, 0, stream>>>(x, 32, 32, neWb, 128, neB, hn, 0);
  k_fill<<<g1(256), 256, 0, stream>>>(stats, 256, 0.f);
  k_bn_stats<<<g1((long)N), 128, 0, stream>>>(hn, N, stats);
  k_bn_apply<<<g1(NH), 256, 0, stream>>>(hn, stats, neG, neBe, h, 1.0f / N, NH);

  // ---- layer helpers
  auto gcn = [&](const unsigned short* Wb, const float* bias, int res) {
    k_gemm<4><<<N / 16, 32, 0, stream>>>(h, 128, 128, Wb, 128, nullptr, xw, 0);
    k_fill<<<g1(NH), 256, 0, stream>>>(agg, NH, 0.f);
    k_gcn_scatter<<<g1((long)E * 32), 256, 0, stream>>>(xw, srcI, dstI, dinv, agg, E);
    k_gcn_self<<<g1(NH), 256, 0, stream>>>(agg, xw, invd, bias, hn, NH);
    k_act<<<g1(NH), 256, 0, stream>>>(hn, h, NH, res);
  };
  auto gat = [&](const unsigned short* Wb, const float* as, const float* ad,
                 const float* bias, int res) {
    k_gemm<4><<<N / 16, 32, 0, stream>>>(h, 128, 128, Wb, 128, nullptr, xw, 0);
    k_gat_al<<<g1((long)N * 8), 256, 0, stream>>>(xw, as, ad, als, ald, N);
    k_gat_selfinit<<<g1((long)N * 8), 256, 0, stream>>>(als, ald, emax, N);
    k_gat_emax<<<g1((long)E * 8), 256, 0, stream>>>(srcI, dstI, als, ald, emax, E);
    k_gat_denominit<<<g1((long)N * 8), 256, 0, stream>>>(als, ald, emax, denom, N);
    k_gat_denom<<<g1((long)E * 8), 256, 0, stream>>>(srcI, dstI, als, ald, emax, denom, E);
    k_gat_outinit<<<g1(NH), 256, 0, stream>>>(xw, als, ald, emax, denom, bias, hn, NH);
    k_gat_scatter<<<g1((long)E * 32), 256, 0, stream>>>(xw, srcI, dstI, als, ald, emax, denom, hn, E);
    k_act<<<g1(NH), 256, 0, stream>>>(hn, h, NH, res);
  };
  auto ec = [&](const unsigned short* Wd, const unsigned short* Ws, const float* b1,
                const unsigned short* W2b, const float* b2, int res) {
    k_gemm<4><<<N / 16, 32, 0, stream>>>(h, 128, 128, Wd, 128, b1, hA, 0);
    k_gemm<4><<<N / 16, 32, 0, stream>>>(h, 128, 128, Ws, 128, nullptr, hB, 0);
    k_fill_enc<<<g1(NH), 256, 0, stream>>>((unsigned*)agg, NH, -3.0e38f);
    k_ec_edge<<<E / 16, 32, 0, stream>>>(hA, hB, srcI, dstI, W2b, b2, (unsigned*)agg);
    k_act_ec<<<g1(NH), 256, 0, stream>>>((unsigned*)agg, h, NH, res);
  };

  // ---- six message-passing layers
  gcn(gcn0Wb, gcn0B, 0);                               // i=0
  gat(gat1Wb, gat1As, gat1Ad, gat1B, 1);               // i=1 (residual)
  ec(ec2Wd, ec2Ws, ec2B1, ec2W2b, ec2B2, 0);           // i=2
  gcn(gcn3Wb, gcn3B, 1);                               // i=3 (residual)
  gat(gat4Wb, gat4As, gat4Ad, gat4B, 0);               // i=4
  ec(ec5Wd, ec5Ws, ec5B1, ec5W2b, ec5B2, 1);           // i=5 (residual)

  // ---- graph context -> gvec = g2 @ epW1[384:512] + ep_b1
  k_fill<<<g1(256), 256, 0, stream>>>(stats, 256, 0.f);
  k_bn_stats<<<g1((long)N), 128, 0, stream>>>(h, N, stats);
  k_gc<<<1, 256, 0, stream>>>(stats, 1.0f / N, gcW1, gcB1, gcW2, gcB2, epW1, epB1, gvec);

  // ---- edge-predictor node-side precompute: u1 = h@W1[0:128], u2 = h@W1[128:256]
  k_gemm<4><<<N / 16, 32, 0, stream>>>(h, 128, 128, epW1r0, 256, nullptr, u1, 0);
  k_gemm<4><<<N / 16, 32, 0, stream>>>(h, 128, 128, epW1r1, 256, nullptr, u2, 0);

  // ---- fused edge predictor (edge encoder + 3-layer MLP with LNs)
  k_ep<<<E / 16, 32, 0, stream>>>(ea, srcI, dstI, eeWb, eeB, eeG, eeBe,
                                  epW1r2, u1, u2, gvec,
                                  epG1, epBe1, epW2b, epB2, epG2, epBe2,
                                  epW3, epB3, (float*)d_out);
}
